// TemporalGAT_61933428416929
// MI455X (gfx1250) — matrix operand order repaired and checked by reading, not compiled, not run on hardware
//
#include <hip/hip_runtime.h>
#include <hip/hip_bf16.h>
#include <stdint.h>

#define IN_C    64
#define TIME_DIM 32
#define HID     32
#define HEADS   4
#define D1      128      // HEADS*HID
#define INDIM   96       // IN_C + TIME_DIM
#define OUT_C   2
#define NEG     0.2f
#define CDIV(a,b) (((a)+(b)-1)/(b))

typedef __attribute__((ext_vector_type(16))) _Float16 v16h;
typedef __attribute__((ext_vector_type(8)))  float    v8f;

// ---- ordered-uint encoding for float atomicMax over possibly-negative values
__device__ __forceinline__ unsigned ord_key(float f) {
  unsigned u = __float_as_uint(f);
  return (u & 0x80000000u) ? ~u : (u | 0x80000000u);
}
__device__ __forceinline__ float ord_val(unsigned k) {
  unsigned u = (k & 0x80000000u) ? (k ^ 0x80000000u) : ~k;
  return __uint_as_float(u);
}

// ---- init: out0 <- bias0 rows, dout <- bias1, max/sum accumulators <- identity
__global__ void k_init(float* __restrict__ out0, unsigned* __restrict__ m0,
                       float* __restrict__ s0, unsigned* __restrict__ m1,
                       float* __restrict__ s1, float* __restrict__ dout,
                       const float* __restrict__ bias0,
                       const float* __restrict__ bias1, int Nn) {
  int idx = blockIdx.x * blockDim.x + threadIdx.x;
  if (idx < Nn * D1)    out0[idx] = bias0[idx & (D1 - 1)];
  if (idx < Nn * HEADS) { m0[idx] = 0u; s0[idx] = 0.f; }
  if (idx < Nn)         { m1[idx] = 0u; s1[idx] = 0.f; }
  if (idx < Nn * OUT_C) dout[idx] = bias1[idx & 1];
}

// ---- h = [x | W_time[time_step]]  ->  f16, row-major N x 96
__global__ void k_build_h16(const float* __restrict__ x,
                            const int* __restrict__ tstep,
                            const float* __restrict__ Wtime,
                            _Float16* __restrict__ h16, int Nn) {
  int idx = blockIdx.x * blockDim.x + threadIdx.x;
  if (idx >= Nn * INDIM) return;
  int i = idx / INDIM, j = idx - i * INDIM;
  float v = (j < IN_C) ? x[(size_t)i * IN_C + j]
                       : Wtime[(size_t)tstep[i] * TIME_DIM + (j - IN_C)];
  h16[idx] = (_Float16)v;
}

// ---- transpose+convert weights: W[96x128] fp32 -> Wt[128x96] f16 (K contiguous)
__global__ void k_prep_w(const float* __restrict__ Wl, const float* __restrict__ Wr,
                         _Float16* __restrict__ WtL, _Float16* __restrict__ WtR) {
  int idx = blockIdx.x * blockDim.x + threadIdx.x;
  if (idx >= INDIM * D1) return;
  int k = idx / D1, n = idx - k * D1;
  WtL[(size_t)n * INDIM + k] = (_Float16)Wl[idx];
  WtR[(size_t)n * INDIM + k] = (_Float16)Wr[idx];
}

// ---- layer-0 GEMM via WMMA: one wave computes a 16x16 tile of xl or xr
// A frag (16-bit A 16x32, contiguous k order used here): lane m = lane&15, hi = lane>>4,
//   vgpr q: K = kb + 2q + half + 16*hi
// B frag (32x16): lane n = lane&15, hi = lane>>4, vgpr q: K = kb + 2q + half + 16*hi
__global__ void __launch_bounds__(32)
k_gemm_l0(const _Float16* __restrict__ h16,
          const _Float16* __restrict__ WtL, const _Float16* __restrict__ WtR,
          const float* __restrict__ bL, const float* __restrict__ bR,
          float* __restrict__ xl, float* __restrict__ xr, int Nn) {
  const int lane = threadIdx.x & 31;
  const int lo = lane & 15, hi = lane >> 4;
  const int m0 = blockIdx.x * 16;
  const int n0 = blockIdx.y * 16;
  const int which = blockIdx.z;

  const _Float16* __restrict__ Wt   = which ? WtR : WtL;
  const float*    __restrict__ bias = which ? bR : bL;
  float*          __restrict__ out  = which ? xr : xl;

  v8f c;
  {
    float bv = bias[n0 + lo];
#pragma unroll
    for (int r = 0; r < 8; ++r) c[r] = bv;
  }

  int marow = m0 + lo; if (marow >= Nn) marow = Nn - 1;   // clamp, keep EXEC full
  const _Float16* arow = h16 + (size_t)marow * INDIM + 16 * hi;
  const _Float16* brow = Wt + (size_t)(n0 + lo) * INDIM + 16 * hi;

#pragma unroll
  for (int kb = 0; kb < INDIM; kb += 32) {
    union { v16h v; uint32_t u[8]; } A, B;
#pragma unroll
    for (int q = 0; q < 8; ++q) {
      int kofs = 2 * q;
      A.u[q] = *(const uint32_t*)(arow + kb + kofs);
    }
#pragma unroll
    for (int q = 0; q < 8; ++q)
      B.u[q] = *(const uint32_t*)(brow + kb + 2 * q);
    c = __builtin_amdgcn_wmma_f32_16x16x32_f16(false, A.v, false, B.v,
                                               (short)0, c, false, false);
  }

  // C/D layout: lane holds rows (r + 8*hi), column n0+lo. m0 is block-uniform,
  // so the full-tile test is a single scalar branch; full path is 8 straight
  // global_store_b32 with immediate offsets (r*D1*4 bytes) off one base.
  float* orow = out + (size_t)(m0 + 8 * hi) * D1 + n0 + lo;
  if (m0 + 16 <= Nn) {
#pragma unroll
    for (int r = 0; r < 8; ++r) orow[(size_t)r * D1] = c[r];
  } else {
#pragma unroll
    for (int r = 0; r < 8; ++r) {
      if (m0 + 8 * hi + r < Nn) orow[(size_t)r * D1] = c[r];
    }
  }
}

// ---- layer-0 edge pass A: logits + segment max (thread = (edge, head))
__global__ void k_logits0(const int* __restrict__ ei,
                          const float* __restrict__ xl, const float* __restrict__ xr,
                          const float* __restrict__ att,
                          float* __restrict__ logit, unsigned* __restrict__ mmax,
                          int E, int Nn) {
  int tid = blockIdx.x * blockDim.x + threadIdx.x;
  int ET = E + Nn;
  if (tid >= ET * HEADS) return;
  int e = tid >> 2, h = tid & 3;
  int s, d;
  if (e < E) { s = ei[e]; d = ei[E + e]; } else { s = d = e - E; }
  const float4* a4 = (const float4*)(xl + (size_t)s * D1 + h * HID);
  const float4* b4 = (const float4*)(xr + (size_t)d * D1 + h * HID);
  const float4* t4 = (const float4*)(att + h * HID);
  float acc = 0.f;
#pragma unroll
  for (int q = 0; q < 8; ++q) {
    float4 av = a4[q], bv = b4[q], tv = t4[q];
    float v0 = av.x + bv.x; v0 = v0 > 0.f ? v0 : v0 * NEG;
    float v1 = av.y + bv.y; v1 = v1 > 0.f ? v1 : v1 * NEG;
    float v2 = av.z + bv.z; v2 = v2 > 0.f ? v2 : v2 * NEG;
    float v3 = av.w + bv.w; v3 = v3 > 0.f ? v3 : v3 * NEG;
    acc += v0 * tv.x + v1 * tv.y + v2 * tv.z + v3 * tv.w;
  }
  logit[tid] = acc;
  atomicMax(mmax + (size_t)d * HEADS + h, ord_key(acc));
}

// ---- layer-0 edge pass B: a = exp(logit - m[dst]); segment sum
__global__ void k_expsum0(const int* __restrict__ ei, const unsigned* __restrict__ mmax,
                          float* __restrict__ logit, float* __restrict__ ssum,
                          int E, int Nn) {
  int tid = blockIdx.x * blockDim.x + threadIdx.x;
  int ET = E + Nn;
  if (tid >= ET * HEADS) return;
  int e = tid >> 2, h = tid & 3;
  int d = (e < E) ? ei[E + e] : (e - E);
  float m = ord_val(mmax[(size_t)d * HEADS + h]);
  float a = __expf(logit[tid] - m);
  logit[tid] = a;                                  // reuse buffer: store numerator
  atomicAdd(ssum + (size_t)d * HEADS + h, a);
}

// ---- layer-0 edge pass C: out[dst] += alpha * xl[src]
__global__ void k_agg0(const int* __restrict__ ei, const float* __restrict__ logit,
                       const float* __restrict__ ssum, const float* __restrict__ xl,
                       float* __restrict__ out0, int E, int Nn) {
  int tid = blockIdx.x * blockDim.x + threadIdx.x;
  int ET = E + Nn;
  if (tid >= ET * HEADS) return;
  int e = tid >> 2, h = tid & 3;
  int s, d;
  if (e < E) { s = ei[e]; d = ei[E + e]; } else { s = d = e - E; }
  float alpha = logit[tid] / (ssum[(size_t)d * HEADS + h] + 1e-16f);
  const float4* s4 = (const float4*)(xl + (size_t)s * D1 + h * HID);
  float* dstp = out0 + (size_t)d * D1 + h * HID;
#pragma unroll
  for (int q = 0; q < 8; ++q) {                    // b128 loads, then 4 atomics
    float4 sv = s4[q];
    atomicAdd(dstp + 4 * q + 0, alpha * sv.x);
    atomicAdd(dstp + 4 * q + 1, alpha * sv.y);
    atomicAdd(dstp + 4 * q + 2, alpha * sv.z);
    atomicAdd(dstp + 4 * q + 3, alpha * sv.w);
  }
}

__global__ void k_relu(float* __restrict__ a, int n) {
  int i = blockIdx.x * blockDim.x + threadIdx.x;
  if (i < n) a[i] = fmaxf(a[i], 0.f);
}

// ---- layer-1 transforms (128 -> 2, both W_l1 and W_r1); thread = (node, 4 outputs)
__global__ void k_lin1(const float* __restrict__ h1,
                       const float* __restrict__ Wl1, const float* __restrict__ bl1,
                       const float* __restrict__ Wr1, const float* __restrict__ br1,
                       float* __restrict__ xl1, float* __restrict__ xr1, int Nn) {
  int tid = blockIdx.x * blockDim.x + threadIdx.x;
  if (tid >= Nn * 4) return;
  int i = tid >> 2;
  int j = tid & 1;
  int which = (tid >> 1) & 1;
  const float* W = which ? Wr1 : Wl1;
  const float* hrow = h1 + (size_t)i * D1;
  float acc = which ? br1[j] : bl1[j];
#pragma unroll 8
  for (int k = 0; k < D1; ++k) acc += hrow[k] * W[k * 2 + j];
  (which ? xr1 : xl1)[(size_t)i * 2 + j] = acc;
}

// ---- layer-1 edge passes (1 head, 2 channels); thread = edge
__global__ void k_logits1(const int* __restrict__ ei,
                          const float* __restrict__ xl1, const float* __restrict__ xr1,
                          const float* __restrict__ att1,
                          float* __restrict__ logit, unsigned* __restrict__ mmax,
                          int E, int Nn) {
  int e = blockIdx.x * blockDim.x + threadIdx.x;
  int ET = E + Nn;
  if (e >= ET) return;
  int s, d;
  if (e < E) { s = ei[e]; d = ei[E + e]; } else { s = d = e - E; }
  float acc = 0.f;
#pragma unroll
  for (int c = 0; c < OUT_C; ++c) {
    float v = xl1[(size_t)s * 2 + c] + xr1[(size_t)d * 2 + c];
    v = v > 0.f ? v : v * NEG;
    acc += v * att1[c];
  }
  logit[e] = acc;
  atomicMax(mmax + d, ord_key(acc));
}

__global__ void k_expsum1(const int* __restrict__ ei, const unsigned* __restrict__ mmax,
                          float* __restrict__ logit, float* __restrict__ ssum,
                          int E, int Nn) {
  int e = blockIdx.x * blockDim.x + threadIdx.x;
  int ET = E + Nn;
  if (e >= ET) return;
  int d = (e < E) ? ei[E + e] : (e - E);
  float a = __expf(logit[e] - ord_val(mmax[d]));
  logit[e] = a;
  atomicAdd(ssum + d, a);
}

__global__ void k_agg1(const int* __restrict__ ei, const float* __restrict__ logit,
                       const float* __restrict__ ssum, const float* __restrict__ xl1,
                       float* __restrict__ dout, int E, int Nn) {
  int e = blockIdx.x * blockDim.x + threadIdx.x;
  int ET = E + Nn;
  if (e >= ET) return;
  int s, d;
  if (e < E) { s = ei[e]; d = ei[E + e]; } else { s = d = e - E; }
  float alpha = logit[e] / (ssum[d] + 1e-16f);
  atomicAdd(dout + (size_t)d * 2 + 0, alpha * xl1[(size_t)s * 2 + 0]);
  atomicAdd(dout + (size_t)d * 2 + 1, alpha * xl1[(size_t)s * 2 + 1]);
}

extern "C" void kernel_launch(void* const* d_in, const int* in_sizes, int n_in,
                              void* d_out, int out_size, void* d_ws, size_t ws_size,
                              hipStream_t stream) {
  const float* x      = (const float*)d_in[0];
  const int*   ei     = (const int*)  d_in[1];
  const int*   tstep  = (const int*)  d_in[2];
  const float* Wtime  = (const float*)d_in[3];
  const float* Wl0    = (const float*)d_in[4];
  const float* bl0    = (const float*)d_in[5];
  const float* Wr0    = (const float*)d_in[6];
  const float* br0    = (const float*)d_in[7];
  const float* att0   = (const float*)d_in[8];
  const float* bias0  = (const float*)d_in[9];
  const float* Wl1    = (const float*)d_in[10];
  const float* bl1    = (const float*)d_in[11];
  const float* Wr1    = (const float*)d_in[12];
  const float* br1    = (const float*)d_in[13];
  const float* att1   = (const float*)d_in[14];
  const float* bias1  = (const float*)d_in[15];
  float* dout = (float*)d_out;

  const int N  = in_sizes[0] / IN_C;
  const int E  = in_sizes[1] / 2;
  const int ET = E + N;

  char* ws = (char*)d_ws;
  size_t off = 0;
  auto take = [&](size_t bytes) -> void* {
    void* p = ws + off;
    off += (bytes + 255) & ~(size_t)255;
    return p;
  };
  _Float16* h16   = (_Float16*)take((size_t)N * INDIM * sizeof(_Float16));
  _Float16* WtL   = (_Float16*)take((size_t)D1 * INDIM * sizeof(_Float16));
  _Float16* WtR   = (_Float16*)take((size_t)D1 * INDIM * sizeof(_Float16));
  float*    xl0   = (float*)   take((size_t)N * D1 * sizeof(float));
  float*    xr0   = (float*)   take((size_t)N * D1 * sizeof(float));
  float*    logit0= (float*)   take((size_t)ET * HEADS * sizeof(float));
  unsigned* m0    = (unsigned*)take((size_t)N * HEADS * sizeof(unsigned));
  float*    s0    = (float*)   take((size_t)N * HEADS * sizeof(float));
  float*    out0  = (float*)   take((size_t)N * D1 * sizeof(float));
  float*    xl1   = (float*)   take((size_t)N * OUT_C * sizeof(float));
  float*    xr1   = (float*)   take((size_t)N * OUT_C * sizeof(float));
  unsigned* m1    = (unsigned*)take((size_t)N * sizeof(unsigned));
  float*    s1    = (float*)   take((size_t)N * sizeof(float));
  float*    logit1= logit0;                       // reuse (layer-0 logits dead by then)

  const int T = 256;

  k_init<<<CDIV(N * D1, T), T, 0, stream>>>(out0, m0, s0, m1, s1, dout, bias0, bias1, N);
  k_build_h16<<<CDIV(N * INDIM, T), T, 0, stream>>>(x, tstep, Wtime, h16, N);
  k_prep_w<<<CDIV(INDIM * D1, T), T, 0, stream>>>(Wl0, Wr0, WtL, WtR);

  dim3 ggrid(CDIV(N, 16), D1 / 16, 2);            // 3125 x 8 x 2 waves
  k_gemm_l0<<<ggrid, 32, 0, stream>>>(h16, WtL, WtR, bl0, br0, xl0, xr0, N);

  k_logits0<<<CDIV(ET * HEADS, T), T, 0, stream>>>(ei, xl0, xr0, att0, logit0, m0, E, N);
  k_expsum0<<<CDIV(ET * HEADS, T), T, 0, stream>>>(ei, m0, logit0, s0, E, N);
  k_agg0   <<<CDIV(ET * HEADS, T), T, 0, stream>>>(ei, logit0, s0, xl0, out0, E, N);

  k_relu<<<CDIV(N * D1, T), T, 0, stream>>>(out0, N * D1);
  k_lin1<<<CDIV(N * 4, T), T, 0, stream>>>(out0, Wl1, bl1, Wr1, br1, xl1, xr1, N);

  k_logits1<<<CDIV(ET, T), T, 0, stream>>>(ei, xl1, xr1, att1, logit1, m1, E, N);
  k_expsum1<<<CDIV(ET, T), T, 0, stream>>>(ei, m1, logit1, s1, E, N);
  k_agg1   <<<CDIV(ET, T), T, 0, stream>>>(ei, logit1, s1, xl1, dout, E, N);
}
